// ParallelLinearQuantized_14791867367661
// MI455X (gfx1250) — matrix operand order repaired and checked by reading, not compiled, not run on hardware
//
#include <hip/hip_runtime.h>
#include <hip/hip_bf16.h>

// ParallelLinearQuantized: out = x @ dequant(wq, per-group scales) + bias
//   x: [4,4096] f32 | wq: [4096,16384] int32(int8 values) | scales: [32,16384] f32
//   bias: [16384] f32 | out: [4,16384] f32
//
// HBM-bound: 256 MB weight stream @ 23.3 TB/s => ~11 us floor.
// v_wmma_f32_16x16x32_f16 does all MACs; VALU only dequantizes (int8 exact in f16).
// x is pre-converted to f16 once (device buffer) so the A operand is 2 plain b128 loads.

#define IN_F   4096
#define OUT_F  16384
#define GROUPK 128
#define BATCH  4
#define KSPLIT 8
#define TILE_N 32   // two 16-col WMMA tiles per wave, sharing one A fragment

typedef __attribute__((ext_vector_type(16))) _Float16 v16h;
typedef __attribute__((ext_vector_type(8)))  _Float16 v8h;
typedef __attribute__((ext_vector_type(8)))  float    v8f;

// x in f16, rewritten deterministically at the start of every launch
__device__ __align__(16) _Float16 g_xf16[BATCH * IN_F];

__global__ __launch_bounds__(256)
void x_to_f16_kernel(const float* __restrict__ x) {
    const int i = blockIdx.x * 256 + threadIdx.x;
    if (i < BATCH * IN_F) g_xf16[i] = (_Float16)x[i];
}

__global__ __launch_bounds__(32)
void qlin_wmma_kernel(const int*   __restrict__ wq,
                      const float* __restrict__ scales,
                      float*       __restrict__ part,   // [ksplit][BATCH][OUT_F]
                      int ksplit)
{
    const int tile  = blockIdx.x;          // 32-column output tile
    const int chunk = blockIdx.y;          // K chunk
    const int lane  = threadIdx.x;         // wave32
    const int col   = lane & 15;           // N index within 16-col subtile
    const int hi    = lane >> 4;           // lane half (K-phase selector)
    const int obase = tile * TILE_N;

    const int klen    = IN_F / ksplit;
    const int k0      = chunk * klen;
    const int ngroups = klen / GROUPK;

    v8f acc0 = {}, acc1 = {};

    for (int g = 0; g < ngroups; ++g) {
        const int kg  = k0 + g * GROUPK;
        const int grp = kg / GROUPK;       // global quant-group index
        const float s0 = scales[(size_t)grp * OUT_F + obase + col];
        const float s1 = scales[(size_t)grp * OUT_F + obase + 16 + col];

        v8f c0 = {}, c1 = {};
        #pragma unroll
        for (int st = 0; st < GROUPK / 32; ++st) {
            const int kb = kg + st * 32;

            // ---- A operand (shared by both subtiles): pre-converted f16 x ----
            // A 16x32 order used here (contiguous k): lanes 0-15 row M=lane, elems0-15 = K kb+0..15;
            // lanes 16-31 same rows, K kb+16..31.
            v16h a = {};
            if (col < BATCH) {
                const v8h* xp = reinterpret_cast<const v8h*>(
                    g_xf16 + (size_t)col * IN_F + kb + hi * 16);
                const v8h lo8 = xp[0];     // 16B aligned b128
                const v8h hi8 = xp[1];     // +8 halves
                #pragma unroll
                for (int e = 0; e < 8; ++e) { a[e] = lo8[e]; a[8 + e] = hi8[e]; }
            }

            // ---- B operands: two 16-col subtiles, int8->f16 exact dequant ----
            // B 32x16 order used here (contiguous k, same as A): lane n holds column n; lanes 0-15 K kb+0..15,
            // lanes 16-31 K kb+16..31; VGPR j packs K 2j, 2j+1.
            const int  kbb = kb + hi * 16;
            const int* wp  = wq + (size_t)kbb * OUT_F + obase + col;
            v16h b0, b1;
            #pragma unroll
            for (int j = 0; j < 8; ++j) {
                const int w00 = wp[(size_t)(2 * j)     * OUT_F];
                const int w01 = wp[(size_t)(2 * j + 1) * OUT_F];
                const int w10 = wp[(size_t)(2 * j)     * OUT_F + 16];
                const int w11 = wp[(size_t)(2 * j + 1) * OUT_F + 16];
                b0[2 * j] = (_Float16)w00;  b0[2 * j + 1] = (_Float16)w01;
                b1[2 * j] = (_Float16)w10;  b1[2 * j + 1] = (_Float16)w11;
            }

            c0 = __builtin_amdgcn_wmma_f32_16x16x32_f16(
                    false, a, false, b0, (short)0, c0, false, false);
            c1 = __builtin_amdgcn_wmma_f32_16x16x32_f16(
                    false, a, false, b1, (short)0, c1, false, false);
        }

        // per-group dequant scale applied once per 128-K group
        #pragma unroll
        for (int r = 0; r < 8; ++r) {
            acc0[r] += s0 * c0[r];
            acc1[r] += s1 * c1[r];
        }
    }

    // C/D layout: VGPR r, lanes 0-15 hold M=r (batch r), N=lane.
    if (hi == 0) {
        #pragma unroll
        for (int r = 0; r < BATCH; ++r) {
            part[((size_t)chunk * BATCH + r) * OUT_F + obase + col]      = acc0[r];
            part[((size_t)chunk * BATCH + r) * OUT_F + obase + 16 + col] = acc1[r];
        }
    }
}

__global__ __launch_bounds__(256)
void qlin_reduce_bias(const float* __restrict__ part,
                      const float* __restrict__ bias,
                      float*       __restrict__ out,
                      int ksplit)
{
    const int i = blockIdx.x * blockDim.x + threadIdx.x;   // over BATCH*OUT_F
    if (i < BATCH * OUT_F) {
        const int o = i & (OUT_F - 1);
        float v = bias[o];
        for (int c = 0; c < ksplit; ++c)
            v += part[(size_t)c * BATCH * OUT_F + i];
        out[i] = v;
    }
}

extern "C" void kernel_launch(void* const* d_in, const int* in_sizes, int n_in,
                              void* d_out, int out_size, void* d_ws, size_t ws_size,
                              hipStream_t stream) {
    const float* x      = (const float*)d_in[0];
    const int*   wq     = (const int*)  d_in[1];
    const float* scales = (const float*)d_in[2];
    const float* bias   = (const float*)d_in[3];
    // d_in[4] = group_size scalar (hardcoded 128)
    float* out = (float*)d_out;

    // one-shot: x -> f16 (deterministic rewrite every call)
    x_to_f16_kernel<<<(BATCH * IN_F + 255) / 256, 256, 0, stream>>>(x);

    const size_t need = (size_t)KSPLIT * BATCH * OUT_F * sizeof(float);
    if (ws_size >= need) {
        float* part = (float*)d_ws;
        qlin_wmma_kernel<<<dim3(OUT_F / TILE_N, KSPLIT), 32, 0, stream>>>(
            wq, scales, part, KSPLIT);
        qlin_reduce_bias<<<(BATCH * OUT_F + 255) / 256, 256, 0, stream>>>(
            part, bias, out, KSPLIT);
    } else {
        // fallback: no K-split; use d_out as the (single) partial buffer
        qlin_wmma_kernel<<<dim3(OUT_F / TILE_N, 1), 32, 0, stream>>>(
            wq, scales, out, 1);
        qlin_reduce_bias<<<(BATCH * OUT_F + 255) / 256, 256, 0, stream>>>(
            out, bias, out, 1);
    }
}
